// LMModel_55731495633306
// MI455X (gfx1250) — compile-verified, matrix operand order checked by reading, not run on hardware
//
#include <hip/hip_runtime.h>
#include <math.h>

// ---------------------------------------------------------------------------
// Types for CDNA5 WMMA (wave32): v_wmma_f32_16x16x32_bf16
// ---------------------------------------------------------------------------
typedef __attribute__((ext_vector_type(16))) __bf16 v16bf;
typedef __attribute__((ext_vector_type(8)))  __bf16 v8bf;
typedef __attribute__((ext_vector_type(8)))  float  v8f;

// ---------------------------------------------------------------------------
// Problem constants (from the reference)
// ---------------------------------------------------------------------------
#define S_LEN   128
#define BATCH   32
#define NIN     512
#define HD      512
#define LD      3
#define VSZ     1536                    // NIN + 2*HD
#define NVOC    50257
#define NT_DEC_PAD 3144                 // ceil(50257/16)=3142, padded to 4-tile groups
#define NTG_DEC    786                  // 3144/4 N-tile groups
#define KC_V    48                      // VSZ/32 K-chunks
#define KC_DEC  16                      // HD/32 K-chunks
#define GATE_ELEMS ((size_t)(48*32*32*16))            // 786432 bf16 per packed matrix
#define WDEC_ELEMS ((size_t)KC_DEC*NT_DEC_PAD*32*16)  // 25,755,648 bf16
#define KC_STRIDE  ((size_t)(32*32*16))               // 16384 elems per kc step (NT=32)

// Workspace layout (bytes)
#define OFF_X     ((size_t)0)                       //  4,194,304  x bf16 [S*B, 512]
#define OFF_Y     ((size_t)4194304)                 //  4,194,304  y bf16 [S*B, 512]
#define OFF_G     ((size_t)8388608)                 // 18,874,368  12 packed gate mats
#define OFF_WHY   ((size_t)27262976)                //  1,572,864  packed Why
#define OFF_WDEC  ((size_t)28835840)                // 51,511,296  packed Wdec (zero-padded N)
#define OFF_C     ((size_t)80347136)                //    196,608  c-state f32 [3][32][512]

// ---------------------------------------------------------------------------
// Helpers
// ---------------------------------------------------------------------------
__device__ __forceinline__ __bf16 f2bf(float f) { return (__bf16)f; }  // hw cvt, RNE

__device__ __forceinline__ float sigm(float x) { return 1.0f / (1.0f + __expf(-x)); }

// A-fragment (16x32 bf16, row-major source with given element stride).
// ISA layout: lanes 0-15 hold row M=lane, K 0-7 & 16-23; lanes 16-31 row M=lane-16, K 8-15 & 24-31.
__device__ __forceinline__ v16bf load_a16(const __bf16* rowbase, int stride, int lane) {
  const __bf16* p = rowbase + (size_t)(lane & 15) * stride + ((lane >> 4) << 3);
  v8bf lo = *(const v8bf*)(p);
  v8bf hi = *(const v8bf*)(p + 16);
  return __builtin_shufflevector(lo, hi, 0,1,2,3,4,5,6,7,8,9,10,11,12,13,14,15);
}

template <bool RA>
__device__ __forceinline__ v8f wmma_bf16(v16bf a, v16bf b, v8f c) {
  return __builtin_amdgcn_wmma_f32_16x16x32_bf16(
      /*neg_a=*/false, a, /*neg_b=*/false, b,
      /*c_mod=*/(short)0, c, /*reuse_a=*/RA, /*reuse_b=*/false);
}

// Software-pipelined GEMM segment: one A fragment feeds 4 WMMAs (4 matrices at
// stride `gstride` in the packed B buffer); iteration kc+1's loads are issued
// before iteration kc's WMMAs so loads overlap 4 WMMAs of latency.
__device__ __forceinline__ void gemm4_seg(const __bf16* __restrict__ abase, int astride,
                                          const __bf16* __restrict__ bbase, size_t gstride,
                                          int lane, int nkc, v8f* acc) {
  const __bf16* bl = bbase + ((size_t)lane << 4);
  v16bf a  = load_a16(abase, astride, lane);
  v16bf b0 = *(const v16bf*)(bl);
  v16bf b1 = *(const v16bf*)(bl + gstride);
  v16bf b2 = *(const v16bf*)(bl + 2 * gstride);
  v16bf b3 = *(const v16bf*)(bl + 3 * gstride);
  for (int kc = 0; kc < nkc; ++kc) {
    v16bf a_n = a, b0_n = b0, b1_n = b1, b2_n = b2, b3_n = b3;
    if (kc + 1 < nkc) {
      const __bf16* bln = bl + (size_t)(kc + 1) * KC_STRIDE;
      a_n  = load_a16(abase + (kc + 1) * 32, astride, lane);
      b0_n = *(const v16bf*)(bln);
      b1_n = *(const v16bf*)(bln + gstride);
      b2_n = *(const v16bf*)(bln + 2 * gstride);
      b3_n = *(const v16bf*)(bln + 3 * gstride);
    }
    acc[0] = wmma_bf16<true >(a, b0, acc[0]);
    acc[1] = wmma_bf16<true >(a, b1, acc[1]);
    acc[2] = wmma_bf16<true >(a, b2, acc[2]);
    acc[3] = wmma_bf16<false>(a, b3, acc[3]);
    a = a_n; b0 = b0_n; b1 = b1_n; b2 = b2_n; b3 = b3_n;
  }
}

// Pipelined single-matrix GEMM segment (used for the Why projection).
__device__ __forceinline__ void gemm1_seg(const __bf16* __restrict__ abase, int astride,
                                          const __bf16* __restrict__ bbase,
                                          int lane, int nkc, v8f& acc) {
  const __bf16* bl = bbase + ((size_t)lane << 4);
  v16bf a = load_a16(abase, astride, lane);
  v16bf b = *(const v16bf*)(bl);
  for (int kc = 0; kc < nkc; ++kc) {
    v16bf a_n = a, b_n = b;
    if (kc + 1 < nkc) {
      a_n = load_a16(abase + (kc + 1) * 32, astride, lane);
      b_n = *(const v16bf*)(bl + (size_t)(kc + 1) * KC_STRIDE);
    }
    acc = wmma_bf16<false>(a, b, acc);
    a = a_n; b = b_n;
  }
}

// ---------------------------------------------------------------------------
// Prep kernel 1: embedding gather fp32 -> bf16, row-major [S*B, NIN]
// ---------------------------------------------------------------------------
__global__ void embed_kernel(const int* __restrict__ ids, const float* __restrict__ emb,
                             __bf16* __restrict__ xbf, int total) {
  for (int idx = blockIdx.x * blockDim.x + threadIdx.x; idx < total;
       idx += gridDim.x * blockDim.x) {
    int row = idx >> 9;          // NIN == 512
    int k   = idx & 511;
    int id  = ids[row];
    xbf[idx] = f2bf(emb[(size_t)id * NIN + k]);
  }
}

// ---------------------------------------------------------------------------
// Prep kernel 2: pack W[n,k] (out = v @ W^T, so B[k][n] = W[n][k]) into the
// WMMA B-fragment tile layout: ((kc*NT + nt)*32 + lane)*16 contiguous bf16,
// mirroring the 16-bit striping (two 8-element K runs per lane).
// ---------------------------------------------------------------------------
__global__ void pack_b_kernel(const float* __restrict__ W, __bf16* __restrict__ dst,
                              int NT, int K, int nValid, long long total) {
  for (long long idx = blockIdx.x * (long long)blockDim.x + threadIdx.x; idx < total;
       idx += (long long)gridDim.x * blockDim.x) {
    int e    = (int)(idx & 15);
    int lane = (int)((idx >> 4) & 31);
    long long rest = idx >> 9;
    int nt = (int)(rest % NT);
    int kc = (int)(rest / NT);
    int n = nt * 16 + (lane & 15);
    int k = kc * 32 + ((lane >> 4) << 3) + (e & 7) + ((e >> 3) << 4);
    float v = (n < nValid) ? W[(size_t)n * K + k] : 0.0f;
    dst[idx] = f2bf(v);
  }
}

// ---------------------------------------------------------------------------
// LSTM scan: persistent single workgroup, 512 threads = 16 waves (4/SIMD).
//   - Ping-pong h buffers in LDS (2 x 96 KB of 320 KB): layer j reads
//     H_new[j-1] + H_old[j], writes H_new[j]; reads/writes never alias, so
//     each slot runs GEMM->elementwise->write straight through and each layer
//     needs exactly ONE barrier (3 per timestep).
//   - c state in a small fp32 global buffer (192 KB, L2-resident). Each wave
//     only touches its own slots, so same-wave load/store ordering suffices.
//     Keeping c out of VGPRs lets the j/slot loops stay ROLLED: one compact
//     GEMM-pipeline instance, ~130 live VGPRs, no spills.
// ---------------------------------------------------------------------------
__global__ void __launch_bounds__(512)
lstm_scan_kernel(const __bf16* __restrict__ xbf,
                 const __bf16* __restrict__ gatesP,
                 const __bf16* __restrict__ whyP,
                 const float* __restrict__ bi_g,  const float* __restrict__ bf_g,
                 const float* __restrict__ bo_g,  const float* __restrict__ bg_g,
                 const float* __restrict__ pci_g, const float* __restrict__ pcf_g,
                 const float* __restrict__ pco_g,
                 const float* __restrict__ bhy_g,
                 float* __restrict__ cstate,      // [LD][BATCH][HD] f32
                 __bf16* __restrict__ ybf) {
  extern __shared__ __bf16 smem[];   // 2 x [BATCH][VSZ] bf16 = 192 KB
  const int tid  = threadIdx.x;
  const int lane = tid & 31;
  const int wave = tid >> 5;         // 0..15

  __bf16* hbuf0 = smem;
  __bf16* hbuf1 = smem + (size_t)BATCH * VSZ;
  for (int i = tid; i < 2 * BATCH * VSZ; i += 512) smem[i] = f2bf(0.0f);
  __syncthreads();

  __bf16* hold = hbuf0;   // h from previous timestep
  __bf16* hnew = hbuf1;   // h being produced this timestep

  for (int t = 0; t < S_LEN; ++t) {
    #pragma unroll 1
    for (int j = 0; j < LD; ++j) {
      #pragma unroll 1
      for (int s4 = 0; s4 < 4; ++s4) {
        const int tile = wave * 4 + s4;   // 0..63
        const int mt = tile & 1;          // batch tile (0..1)
        const int ht = tile >> 1;         // hidden N-tile (0..31)
        const __bf16* gl = gatesP + (size_t)(j * 4) * GATE_ELEMS + (size_t)ht * 512;
        const __bf16* xrow = xbf + ((size_t)(t * BATCH + mt * 16)) * NIN;

        // ---- GEMM: gate pre-activations for this 16x16 slot ----
        v8f acc[4] = {};
        // segment 1: x_t (kc 0..15, global, L2-resident)
        gemm4_seg(xrow, NIN, gl, GATE_ELEMS, lane, 16, acc);
        // segment 2: h of layer j-1 this timestep (kc 16..31, H_new); zero for j==0
        if (j > 0)
          gemm4_seg(hnew + (size_t)(mt * 16) * VSZ + (j - 1) * HD, VSZ,
                    gl + (size_t)16 * KC_STRIDE, GATE_ELEMS, lane, 16, acc);
        // segment 3: h of layer j previous timestep (kc 32..47, H_old)
        gemm4_seg(hold + (size_t)(mt * 16) * VSZ + j * HD, VSZ,
                  gl + (size_t)32 * KC_STRIDE, GATE_ELEMS, lane, 16, acc);

        // ---- peephole-LSTM elementwise, c update, h write (to H_new only) ----
        const int h = ht * 16 + (lane & 15);
        const int mrow = mt * 16 + ((lane >> 4) << 3);
        const float bi_v  = bi_g [j * HD + h];
        const float bf_v  = bf_g [j * HD + h];
        const float bo_v  = bo_g [j * HD + h];
        const float bg_v  = bg_g [j * HD + h];
        const float pci_v = pci_g[j * HD + h];
        const float pcf_v = pcf_g[j * HD + h];
        const float pco_v = pco_g[j * HD + h];
        float* cbase = cstate + ((size_t)(j * BATCH + mrow)) * HD + h;

        v8f cvec = {};
        if (t > 0) {                       // wave-uniform branch; t==0 -> c = 0
          #pragma unroll
          for (int r = 0; r < 8; ++r) cvec[r] = cbase[(size_t)r * HD];
        }
        v8f hvec;
        #pragma unroll
        for (int r = 0; r < 8; ++r) {
          float c0 = cvec[r];
          float iv = sigm(acc[0][r] + bi_v + pci_v * c0);
          float fv = sigm(acc[1][r] + bf_v + pcf_v * c0);
          float ov = sigm(acc[2][r] + bo_v + pco_v * c0);   // peephole on OLD c
          float gv = tanhf(acc[3][r] + bg_v);
          float cn = iv * gv + fv * c0;
          cvec[r] = cn;
          hvec[r] = ov * tanhf(cn);
        }
        #pragma unroll
        for (int r = 0; r < 8; ++r) cbase[(size_t)r * HD] = cvec[r];
        #pragma unroll
        for (int r = 0; r < 8; ++r)
          hnew[(size_t)(mrow + r) * VSZ + j * HD + h] = f2bf(hvec[r]);
      }
      __syncthreads();   // H_new[block j] visible before layer j+1 reads it
    }

    // ---- y_t = concat(h) [32x1536] @ Why^T + bhy -> bf16 for the decoder ----
    // Reads H_new only; next step's writes target H_old, so no extra barrier.
    #pragma unroll 1
    for (int s4 = 0; s4 < 4; ++s4) {
      const int tile = wave * 4 + s4;
      const int mt = tile & 1;
      const int ht = tile >> 1;
      v8f yacc = {};
      gemm1_seg(hnew + (size_t)(mt * 16) * VSZ, VSZ, whyP + (size_t)ht * 512,
                lane, KC_V, yacc);
      const int h = ht * 16 + (lane & 15);
      const float bhv = bhy_g[h];
      const int mrow = mt * 16 + ((lane >> 4) << 3);
      #pragma unroll
      for (int r = 0; r < 8; ++r)
        ybf[((size_t)(t * BATCH) + mrow + r) * HD + h] = f2bf(yacc[r] + bhv);
    }

    // swap ping-pong buffers: this step's H_new is next step's H_old
    __bf16* tmp = hold; hold = hnew; hnew = tmp;
  }
}

// ---------------------------------------------------------------------------
// Decoder GEMM: logits[4096 x 50257] = y[4096x512] @ Wdec^T + bdec
// Each wave owns a 16x64 output strip (4 N-tiles): one A fragment feeds 4
// WMMAs, pipelined one kc ahead. 8 waves/WG share the N-group's B fragments
// (L2-resident, 51 MB bf16); K=512 in 16 pipelined steps.
// ---------------------------------------------------------------------------
__global__ void __launch_bounds__(256)
dec_kernel(const __bf16* __restrict__ ybf, const __bf16* __restrict__ wdecP,
           const float* __restrict__ bdec, float* __restrict__ out) {
  const int lane = threadIdx.x & 31;
  const int wave = threadIdx.x >> 5;
  const int nt0 = blockIdx.x * 4;            // first of 4 N-tiles
  const int mt  = blockIdx.y * 8 + wave;     // M-tile (0..255)
  const size_t bstep = (size_t)NT_DEC_PAD * 512;  // elems per kc step

  v8f acc[4] = {};
  const __bf16* abase = ybf + (size_t)(mt * 16) * HD;
  const __bf16* bl = wdecP + (size_t)nt0 * 512 + ((size_t)lane << 4);

  v16bf a  = load_a16(abase, HD, lane);
  v16bf b0 = *(const v16bf*)(bl);
  v16bf b1 = *(const v16bf*)(bl + 512);
  v16bf b2 = *(const v16bf*)(bl + 1024);
  v16bf b3 = *(const v16bf*)(bl + 1536);
  for (int kc = 0; kc < KC_DEC; ++kc) {
    v16bf a_n = a, b0_n = b0, b1_n = b1, b2_n = b2, b3_n = b3;
    if (kc + 1 < KC_DEC) {
      const __bf16* bln = bl + (size_t)(kc + 1) * bstep;
      a_n  = load_a16(abase + (kc + 1) * 32, HD, lane);
      b0_n = *(const v16bf*)(bln);
      b1_n = *(const v16bf*)(bln + 512);
      b2_n = *(const v16bf*)(bln + 1024);
      b3_n = *(const v16bf*)(bln + 1536);
    }
    acc[0] = wmma_bf16<true >(a, b0, acc[0]);
    acc[1] = wmma_bf16<true >(a, b1, acc[1]);
    acc[2] = wmma_bf16<true >(a, b2, acc[2]);
    acc[3] = wmma_bf16<false>(a, b3, acc[3]);
    a = a_n; b0 = b0_n; b1 = b1_n; b2 = b2_n; b3 = b3_n;
  }

  const int mrow = mt * 16 + ((lane >> 4) << 3);
  #pragma unroll
  for (int u = 0; u < 4; ++u) {
    const int n = (nt0 + u) * 16 + (lane & 15);
    if (n < NVOC) {
      const float bias = bdec[n];
      #pragma unroll
      for (int r = 0; r < 8; ++r)
        out[(size_t)(mrow + r) * NVOC + n] = acc[u][r] + bias;
    }
  }
}

// ---------------------------------------------------------------------------
// Launch
// ---------------------------------------------------------------------------
extern "C" void kernel_launch(void* const* d_in, const int* in_sizes, int n_in,
                              void* d_out, int out_size, void* d_ws, size_t ws_size,
                              hipStream_t stream) {
  (void)in_sizes; (void)n_in; (void)out_size; (void)ws_size;
  const int*   ids  = (const int*)  d_in[0];
  const float* emb  = (const float*)d_in[1];
  const float* Wi   = (const float*)d_in[2];
  const float* bi   = (const float*)d_in[3];
  const float* Wf   = (const float*)d_in[4];
  const float* bff  = (const float*)d_in[5];
  const float* Wo   = (const float*)d_in[6];
  const float* bo   = (const float*)d_in[7];
  const float* Wg   = (const float*)d_in[8];
  const float* bg   = (const float*)d_in[9];
  const float* pci  = (const float*)d_in[10];
  const float* pcf  = (const float*)d_in[11];
  const float* pco  = (const float*)d_in[12];
  const float* Why  = (const float*)d_in[13];
  const float* bhy  = (const float*)d_in[14];
  const float* Wdec = (const float*)d_in[15];
  const float* bdec = (const float*)d_in[16];
  float* out = (float*)d_out;

  char* ws = (char*)d_ws;
  __bf16* xbf   = (__bf16*)(ws + OFF_X);
  __bf16* ybf   = (__bf16*)(ws + OFF_Y);
  __bf16* gates = (__bf16*)(ws + OFF_G);
  __bf16* whyP  = (__bf16*)(ws + OFF_WHY);
  __bf16* wdecP = (__bf16*)(ws + OFF_WDEC);
  float*  cst   = (float*) (ws + OFF_C);

  // 1) embedding gather -> bf16
  embed_kernel<<<8192, 256, 0, stream>>>(ids, emb, xbf, S_LEN * BATCH * NIN);

  // 2) pack gate weights (12 matrices of [512 x 1536]) into WMMA B layout
  const float* gsrc[4] = {Wi, Wf, Wo, Wg};
  for (int j = 0; j < LD; ++j)
    for (int g = 0; g < 4; ++g)
      pack_b_kernel<<<3072, 256, 0, stream>>>(
          gsrc[g] + (size_t)j * HD * VSZ, gates + (size_t)(j * 4 + g) * GATE_ELEMS,
          /*NT=*/32, /*K=*/VSZ, /*nValid=*/HD, (long long)GATE_ELEMS);

  // 3) pack Why [512 x 1536] and Wdec [50257 x 512] (N zero-padded to 50304)
  pack_b_kernel<<<3072, 256, 0, stream>>>(Why, whyP, 32, VSZ, HD, (long long)GATE_ELEMS);
  pack_b_kernel<<<100608, 256, 0, stream>>>(Wdec, wdecP, NT_DEC_PAD, HD, NVOC,
                                            (long long)WDEC_ELEMS);

  // 4) recurrent scan (persistent WG, 192 KB dynamic LDS ping-pong h buffers)
  lstm_scan_kernel<<<1, 512, 2 * BATCH * VSZ * sizeof(__bf16), stream>>>(
      xbf, gates, whyP, bi, bff, bo, bg, pci, pcf, pco, bhy, cst, ybf);

  // 5) decoder GEMM -> fp32 logits (16x64 strip per wave)
  dec_kernel<<<dim3(NTG_DEC, 32), 256, 0, stream>>>(ybf, wdecP, bdec, out);
}
